// SepsisLSTMModel_26156350832813
// MI455X (gfx1250) — matrix operand order repaired and checked by reading, not compiled, not run on hardware
//
#include <hip/hip_runtime.h>
#include <hip/hip_bf16.h>

// ---- shapes ----
#define TT 512
#define BB 32
#define II 64
#define HH 256
#define DD 512        // 2*H
#define G4H 1024      // 4*H

typedef __attribute__((ext_vector_type(16))) __bf16 v16bf;
typedef __attribute__((ext_vector_type(8)))  float  v8f;

__device__ __forceinline__ float sigf(float x){ return 1.0f/(1.0f+__expf(-x)); }

__device__ __forceinline__ void cvt8(v16bf& v, int base, float4 a, float4 b){
  v[base+0]=(__bf16)a.x; v[base+1]=(__bf16)a.y; v[base+2]=(__bf16)a.z; v[base+3]=(__bf16)a.w;
  v[base+4]=(__bf16)b.x; v[base+5]=(__bf16)b.y; v[base+6]=(__bf16)b.z; v[base+7]=(__bf16)b.w;
}

// ---------------- weight conversion ----------------
__global__ void k_cvt_bf16(const float* __restrict__ src, __bf16* __restrict__ dst, int n){
  int i = blockIdx.x*blockDim.x + threadIdx.x;
  if (i < n) dst[i] = (__bf16)src[i];
}

// dst[p][g] = src[2p][g] + src[2p+1][g]   (rows of 1024)
__global__ void k_bias_sum(const float* __restrict__ src, float* __restrict__ dst, int npairs){
  int i = blockIdx.x*blockDim.x + threadIdx.x;
  if (i < npairs*G4H){
    int p = i >> 10, g = i & (G4H-1);
    dst[i] = src[(size_t)(2*p)*G4H + g] + src[(size_t)(2*p+1)*G4H + g];
  }
}

// ---------------- xg = in @ w_ih^T + (b0+b1), per direction ----------------
// in: (B,T,K) f32, row r=b*T+t contiguous in k.  w: (2,4H,K) bf16.  bsum: (2,4H) f32.
// xg out: (2,T,B,4H) bf16.  One wave per 16x16 tile; blockDim 256 (8 waves).
__global__ void k_xg_gemm(const float* __restrict__ in, int K,
                          const __bf16* __restrict__ w,
                          const float* __restrict__ bsum,
                          __bf16* __restrict__ xg)
{
  const int lane = threadIdx.x & 31;
  const int wave = threadIdx.x >> 5;
  const int MT = (BB*TT)/16;            // 1024
  const int NT = G4H/16;                // 64
  int wid = blockIdx.x*8 + wave;
  int dir = wid / (MT*NT);
  int rem = wid - dir*(MT*NT);
  int mt  = rem / NT;
  int nt  = rem - mt*NT;

  const int hl = lane >> 4;
  const int ln = lane & 15;
  const int ng = nt*16 + ln;

  float bsv = bsum[(size_t)dir*G4H + ng];
  v8f acc;
  #pragma unroll
  for (int v=0; v<8; ++v) acc[v] = bsv;

  const float*  arow = in + (size_t)(mt*16 + ln)*K;
  const __bf16* brow = w + (size_t)dir*G4H*K + (size_t)ng*K;

  for (int kc = 0; kc < K/32; ++kc){
    int klo = kc*32 + 16*hl;
    const float4* p0 = (const float4*)(arow + klo);
    const float4* p1 = (const float4*)(arow + klo + 8);
    v16bf av;
    cvt8(av, 0, p0[0], p0[1]);
    cvt8(av, 8, p1[0], p1[1]);
    v16bf bv = *(const v16bf*)(brow + kc*32 + 16*hl);
    acc = __builtin_amdgcn_wmma_f32_16x16x32_bf16(false, av, false, bv, (short)0, acc, false, false);
  }
  #pragma unroll
  for (int v=0; v<8; ++v){
    int row = mt*16 + v + 8*hl;          // = b*T + t
    int b = row >> 9, t = row & (TT-1);
    xg[ (((size_t)dir*TT + t)*BB + b)*G4H + ng ] = (__bf16)acc[v];
  }
}

// ---------------- LSTM recurrence (one layer, both directions) ----------------
// grid 4: block = (dir<<1)|batchhalf; 512 threads = 16 waves, each wave owns 4 N-tiles.
// xg: (2,T,B,4H) bf16;  whh: (2,4H,H) bf16;  out: (B,T,D) f32 (dir*H column offset).
__global__ void k_lstm_rec(const __bf16* __restrict__ xg,
                           const __bf16* __restrict__ whh,
                           float* __restrict__ out)
{
  __shared__ __bf16 h_sh[16][HH];        // 8 KB
  __shared__ float  c_sh[16][HH];        // 16 KB
  __shared__ float  g_sh[16][G4H];       // 64 KB

  const int tid  = threadIdx.x;
  const int lane = tid & 31;
  const int wave = tid >> 5;             // 0..15
  const int dir  = blockIdx.x >> 1;
  const int bh   = blockIdx.x & 1;
  const int hl   = lane >> 4;
  const int ln   = lane & 15;

  for (int i = tid; i < 16*HH; i += 512){
    (&h_sh[0][0])[i] = (__bf16)0.0f;
    (&c_sh[0][0])[i] = 0.0f;
  }
  __syncthreads();

  const __bf16* wd = whh + (size_t)dir*G4H*HH;

  for (int t = 0; t < TT; ++t){
    const int tg = dir ? (TT-1-t) : t;
    const __bf16* xgt = xg + (((size_t)dir*TT + tg)*BB + bh*16)*G4H;
    if (t+1 < TT){
      const int tn = dir ? (TT-2-t) : (t+1);
      __builtin_prefetch(xg + (((size_t)dir*TT + tn)*BB + bh*16)*G4H + tid*2, 0, 0);
    }

    #pragma unroll
    for (int i = 0; i < 4; ++i){
      const int nt = wave*4 + i;
      const int ng = nt*16 + ln;
      v8f acc;
      #pragma unroll
      for (int v=0; v<8; ++v) acc[v] = (float)xgt[(size_t)(v + 8*hl)*G4H + ng];

      const __bf16* brow = wd + (size_t)ng*HH;
      #pragma unroll
      for (int kc = 0; kc < HH/32; ++kc){
        union { v16bf v; uint4 q[2]; } u;
        u.q[0] = *(const uint4*)&h_sh[ln][kc*32 + 16*hl];
        u.q[1] = *(const uint4*)&h_sh[ln][kc*32 + 16*hl + 8];
        v16bf bv = *(const v16bf*)(brow + kc*32 + 16*hl);
        acc = __builtin_amdgcn_wmma_f32_16x16x32_bf16(false, u.v, false, bv, (short)0, acc, false, false);
      }
      #pragma unroll
      for (int v=0; v<8; ++v) g_sh[v + 8*hl][ng] = acc[v];
    }
    __syncthreads();

    #pragma unroll
    for (int s = 0; s < 8; ++s){
      int idx = s*512 + tid;
      int b = idx >> 8, j = idx & (HH-1);
      float iv = g_sh[b][j];
      float fv = g_sh[b][HH   + j];
      float gv = g_sh[b][2*HH + j];
      float ov = g_sh[b][3*HH + j];
      float c  = sigf(fv)*c_sh[b][j] + sigf(iv)*tanhf(gv);
      float h  = sigf(ov)*tanhf(c);
      c_sh[b][j] = c;
      h_sh[b][j] = (__bf16)h;
      out[(((size_t)(bh*16+b))*TT + tg)*DD + dir*HH + j] = h;
    }
    __syncthreads();
  }
}

// ---------------- post 1: hf = h2*mask + x@proj^T + proj_b ; scores = hf.attn + b ----
__global__ void k_post1(const float* __restrict__ h2, const float* __restrict__ x,
                        const float* __restrict__ mask,
                        const float* __restrict__ projw, const float* __restrict__ projb,
                        const float* __restrict__ attnw, const float* __restrict__ attnb,
                        float* __restrict__ hf, float* __restrict__ scores)
{
  __shared__ float xs[II];
  __shared__ float red[256];
  const int bt = blockIdx.x;             // b*T + t
  const int tid = threadIdx.x;           // 256
  if (tid < II) xs[tid] = x[(size_t)bt*II + tid];
  __syncthreads();
  const float mk = mask[bt];
  float part = 0.0f;
  #pragma unroll
  for (int r = 0; r < 2; ++r){
    int j = tid + r*256;
    float acc = projb[j];
    const float* pw = projw + (size_t)j*II;
    #pragma unroll 8
    for (int k = 0; k < II; ++k) acc += xs[k]*pw[k];
    float v = h2[(size_t)bt*DD + j]*mk + acc;
    hf[(size_t)bt*DD + j] = v;
    part += v*attnw[j];
  }
  red[tid] = part;
  __syncthreads();
  for (int s = 128; s > 0; s >>= 1){
    if (tid < s) red[tid] += red[tid+s];
    __syncthreads();
  }
  if (tid == 0) scores[bt] = red[0] + attnb[0];
}

// ---------------- post 2: causal online-softmax pool -> pooled (bf16) ----------------
__global__ void k_pool(const float* __restrict__ hf, const float* __restrict__ scores,
                       __bf16* __restrict__ pooled)
{
  const int b = blockIdx.x;              // 32
  const int j = threadIdx.x;             // 512
  float m = -1e30f, num = 0.0f, den = 0.0f;
  for (int t = 0; t < TT; ++t){
    float s  = scores[b*TT + t];
    float mn = fmaxf(m, s);
    float a  = __expf(m - mn);
    float e  = __expf(s - mn);
    float ht = hf[((size_t)b*TT + t)*DD + j];
    num = num*a + e*ht;
    den = den*a + e;
    m = mn;
    pooled[((size_t)b*TT + t)*DD + j] = (__bf16)(num/den);
  }
}

// ---------------- post 3: z = relu(pooled @ out1^T + b1), WMMA ----------------
__global__ void k_z_gemm(const __bf16* __restrict__ A,   // (BT,512) bf16
                         const __bf16* __restrict__ Wb,  // (256,512) bf16
                         const float* __restrict__ bias, // (256)
                         float* __restrict__ Z)          // (BT,256)
{
  const int lane = threadIdx.x & 31;
  const int wave = threadIdx.x >> 5;
  const int NT = HH/16;                  // 16
  int wid = blockIdx.x*8 + wave;
  int mt = wid / NT, nt = wid - mt*NT;
  const int hl = lane >> 4, ln = lane & 15;
  const int ng = nt*16 + ln;

  float bsv = bias[ng];
  v8f acc;
  #pragma unroll
  for (int v=0; v<8; ++v) acc[v] = bsv;

  const __bf16* arow = A  + (size_t)(mt*16 + ln)*DD;
  const __bf16* brow = Wb + (size_t)ng*DD;
  #pragma unroll
  for (int kc = 0; kc < DD/32; ++kc){
    union { v16bf v; uint4 q[2]; } u;
    u.q[0] = *(const uint4*)(arow + kc*32 + 16*hl);
    u.q[1] = *(const uint4*)(arow + kc*32 + 16*hl + 8);
    v16bf bv = *(const v16bf*)(brow + kc*32 + 16*hl);
    acc = __builtin_amdgcn_wmma_f32_16x16x32_bf16(false, u.v, false, bv, (short)0, acc, false, false);
  }
  #pragma unroll
  for (int v=0; v<8; ++v){
    float z = acc[v];
    Z[(size_t)(mt*16 + v + 8*hl)*HH + ng] = z > 0.0f ? z : 0.0f;
  }
}

// ---------------- post 4: out = sigmoid(z @ out2^T + b2) ----------------
__global__ void k_out(const float* __restrict__ Z, const float* __restrict__ w2,
                      const float* __restrict__ b2, float* __restrict__ out)
{
  int i = blockIdx.x*blockDim.x + threadIdx.x;
  if (i >= BB*TT) return;
  float acc = b2[0];
  const float* zr = Z + (size_t)i*HH;
  #pragma unroll 8
  for (int k = 0; k < HH; ++k) acc += zr[k]*w2[k];
  out[i] = 1.0f/(1.0f+__expf(-acc));
}

// ---------------- launcher ----------------
extern "C" void kernel_launch(void* const* d_in, const int* in_sizes, int n_in,
                              void* d_out, int out_size, void* d_ws, size_t ws_size,
                              hipStream_t stream)
{
  const float* x      = (const float*)d_in[0];
  const float* mask   = (const float*)d_in[1];
  const float* w_ih0  = (const float*)d_in[2];
  const float* w_hh0  = (const float*)d_in[3];
  const float* b0     = (const float*)d_in[4];
  const float* w_ih12 = (const float*)d_in[5];
  const float* w_hh12 = (const float*)d_in[6];
  const float* b12    = (const float*)d_in[7];
  const float* attnw  = (const float*)d_in[8];
  const float* attnb  = (const float*)d_in[9];
  const float* projw  = (const float*)d_in[10];
  const float* projb  = (const float*)d_in[11];
  const float* out1w  = (const float*)d_in[12];
  const float* out1b  = (const float*)d_in[13];
  const float* out2w  = (const float*)d_in[14];
  const float* out2b  = (const float*)d_in[15];
  float* out = (float*)d_out;

  char* ws = (char*)d_ws;
  const size_t MB32 = 33554432;
  float*  bufA   = (float*)(ws);                       // (B,T,D) f32
  float*  bufB   = (float*)(ws + MB32);                // (B,T,D) f32
  __bf16* xg     = (__bf16*)(ws + 2*MB32);             // (2,T,B,4H) bf16, 64MB
  __bf16* pooled = (__bf16*)(ws + 2*MB32);             // overlay (B,T,D) bf16, 16MB
  float*  zbuf   = (float*)(ws + 2*MB32 + 16777216);   // overlay (B,T,H) f32, 16MB
  float*  scores = (float*)(ws + 2*MB32 + 2*16777216); // overlay (B,T) f32

  size_t wo = 134217728;                               // 128MB: weight region
  __bf16* wih0_bf  = (__bf16*)(ws + wo); wo += (size_t)2*G4H*II*2;
  __bf16* whh0_bf  = (__bf16*)(ws + wo); wo += (size_t)2*G4H*HH*2;
  __bf16* wih12_bf = (__bf16*)(ws + wo); wo += (size_t)4*G4H*DD*2;
  __bf16* whh12_bf = (__bf16*)(ws + wo); wo += (size_t)4*G4H*HH*2;
  __bf16* out1_bf  = (__bf16*)(ws + wo); wo += (size_t)HH*DD*2;
  float*  bsum0    = (float*)(ws + wo);  wo += (size_t)2*G4H*4;
  float*  bsum12   = (float*)(ws + wo);  wo += (size_t)4*G4H*4;

  // --- convert weights to bf16, pre-sum biases ---
  auto cvt = [&](const float* s, __bf16* d, int n){
    k_cvt_bf16<<<(n+255)/256, 256, 0, stream>>>(s, d, n);
  };
  cvt(w_ih0,  wih0_bf,  2*G4H*II);
  cvt(w_hh0,  whh0_bf,  2*G4H*HH);
  cvt(w_ih12, wih12_bf, 4*G4H*DD);
  cvt(w_hh12, whh12_bf, 4*G4H*HH);
  cvt(out1w,  out1_bf,  HH*DD);
  k_bias_sum<<<(2*G4H+255)/256, 256, 0, stream>>>(b0,  bsum0,  2);
  k_bias_sum<<<(4*G4H+255)/256, 256, 0, stream>>>(b12, bsum12, 4);

  const int xg_blocks = (2*(BB*TT/16)*(G4H/16))/8;     // 16384

  // --- layer 0: x (K=64) -> bufA ---
  k_xg_gemm<<<xg_blocks, 256, 0, stream>>>(x, II, wih0_bf, bsum0, xg);
  k_lstm_rec<<<4, 512, 0, stream>>>(xg, whh0_bf, bufA);

  // --- layer 1: bufA (K=512) -> bufB ---
  k_xg_gemm<<<xg_blocks, 256, 0, stream>>>(bufA, DD, wih12_bf, bsum12, xg);
  k_lstm_rec<<<4, 512, 0, stream>>>(xg, whh12_bf, bufB);

  // --- layer 2: bufB (K=512) -> bufA ---
  k_xg_gemm<<<xg_blocks, 256, 0, stream>>>(bufB, DD, wih12_bf + (size_t)2*G4H*DD,
                                           bsum12 + 2*G4H, xg);
  k_lstm_rec<<<4, 512, 0, stream>>>(xg, whh12_bf + (size_t)2*G4H*HH, bufA);

  // --- post ---
  k_post1<<<BB*TT, 256, 0, stream>>>(bufA, x, mask, projw, projb, attnw, attnb, bufB, scores);
  k_pool<<<BB, 512, 0, stream>>>(bufB, scores, pooled);
  k_z_gemm<<<((BB*TT/16)*(HH/16))/8, 256, 0, stream>>>(pooled, out1_bf, out1b, zbuf);
  k_out<<<(BB*TT+255)/256, 256, 0, stream>>>(zbuf, out2w, out2b, out);
}
